// HSICIndependenceLoss_67954972557313
// MI455X (gfx1250) — compile-verified, matrix operand order checked by reading, not run on hardware
//
#include <hip/hip_runtime.h>
#include <hip/hip_bf16.h>
#include <stdint.h>

typedef __attribute__((ext_vector_type(16))) __bf16 v16bf;
typedef __attribute__((ext_vector_type(8)))  __bf16 v8bf;
typedef __attribute__((ext_vector_type(8)))  float  v8f;

#define NPTS 4096
#define DZ   512
#define DN   128

__device__ __forceinline__ unsigned short f32_to_bf16_rne(float f) {
  unsigned u = __float_as_uint(f);
  u = u + 0x7FFFu + ((u >> 16) & 1u);
  return (unsigned short)(u >> 16);
}
__device__ __forceinline__ float bf16_to_f32(unsigned short h) {
  return __uint_as_float(((unsigned)h) << 16);
}
__device__ __forceinline__ unsigned sortable_key(float f) {
  unsigned u = __float_as_uint(f);
  return (u & 0x80000000u) ? ~u : (u | 0x80000000u);   // monotonic float -> uint
}

// Load the 16 bf16 A/B-operand elements for this lane: two contiguous 8-element
// (16-byte) runs at +0 and +16 elements (ISA 7.12.2 16-bit operand layout).
__device__ __forceinline__ v16bf load_op16(const __bf16* p) {
  v8bf lo = *(const v8bf*)p;
  v8bf hi = *(const v8bf*)(p + 16);
  return __builtin_shufflevector(lo, hi, 0,1,2,3,4,5,6,7,8,9,10,11,12,13,14,15);
}

// ---------------- split f32 -> (hi, lo) bf16 pair ----------------
__global__ void split_bf16_kernel(const float* __restrict__ X,
                                  unsigned short* __restrict__ hi,
                                  unsigned short* __restrict__ lo,
                                  int total) {
  int idx = blockIdx.x * blockDim.x + threadIdx.x;
  if (idx >= total) return;
  float v = X[idx];
  unsigned short h = f32_to_bf16_rne(v);
  hi[idx] = h;
  lo[idx] = f32_to_bf16_rne(v - bf16_to_f32(h));
}

// ---------------- row squared norms ----------------
__global__ void row_sqnorm_kernel(const float* __restrict__ X,
                                  float* __restrict__ x2, int d) {
  __shared__ float sh[256];
  const int i = blockIdx.x;
  float s = 0.f;
  for (int k = threadIdx.x; k < d; k += 256) {
    float v = X[(size_t)i * d + k];
    s += v * v;
  }
  sh[threadIdx.x] = s; __syncthreads();
  for (int off = 128; off > 0; off >>= 1) {
    if (threadIdx.x < off) sh[threadIdx.x] += sh[threadIdx.x + off];
    __syncthreads();
  }
  if (threadIdx.x == 0) x2[i] = sh[0];
}

// ---------------- d2 = x2_i + x2_j - 2*X X^T  (bf16x3 WMMA, 2x2 tiles/wave) ----
// 8 waves/block, each wave owns a 32x32 macro-tile = 2x2 WMMA tiles (4 accums).
// Per k-step: 8 operand vectors feed 12 v_wmma_f32_16x16x32_bf16 (hi*hi+hi*lo+lo*hi).
// Epilogue fuses radix-select pass 1: LDS histogram of the top byte of the
// sortable key of every produced d2 value (integer atomics -> deterministic).
__global__ void __launch_bounds__(256) d2_wmma_hist_kernel(
    const unsigned short* __restrict__ Xhi_u,
    const unsigned short* __restrict__ Xlo_u,
    const float* __restrict__ x2,
    float* __restrict__ d2,
    int n, int d,
    unsigned* __restrict__ hist) {
  __shared__ unsigned lh[256];
  lh[threadIdx.x] = 0u;
  __syncthreads();

  const int wave = threadIdx.x >> 5;
  const int lane = threadIdx.x & 31;
  const int mtpr = n >> 5;                     // 32-wide macro tiles per row
  const int tile = blockIdx.x * 8 + wave;
  const int ti = tile / mtpr;
  const int tj = tile % mtpr;
  const int i0 = ti << 5;
  const int j0 = tj << 5;
  const int mrow = lane & 15;
  const int kb = (lane < 16) ? 0 : 8;

  const __bf16* A0h = (const __bf16*)Xhi_u + (size_t)(i0 + mrow) * d;
  const __bf16* A0l = (const __bf16*)Xlo_u + (size_t)(i0 + mrow) * d;
  const __bf16* A1h = A0h + (size_t)16 * d;
  const __bf16* A1l = A0l + (size_t)16 * d;
  const __bf16* B0h = (const __bf16*)Xhi_u + (size_t)(j0 + mrow) * d;
  const __bf16* B0l = (const __bf16*)Xlo_u + (size_t)(j0 + mrow) * d;
  const __bf16* B1h = B0h + (size_t)16 * d;
  const __bf16* B1l = B0l + (size_t)16 * d;

  v8f acc00 = {}, acc01 = {}, acc10 = {}, acc11 = {};
  for (int k0 = 0; k0 < d; k0 += 32) {
    const int ko = k0 + kb;
    v16bf a0h = load_op16(A0h + ko);
    v16bf a0l = load_op16(A0l + ko);
    v16bf a1h = load_op16(A1h + ko);
    v16bf a1l = load_op16(A1l + ko);
    v16bf b0h = load_op16(B0h + ko);
    v16bf b0l = load_op16(B0l + ko);
    v16bf b1h = load_op16(B1h + ko);
    v16bf b1l = load_op16(B1l + ko);

    acc00 = __builtin_amdgcn_wmma_f32_16x16x32_bf16(false, a0h, false, b0h, (short)0, acc00, false, false);
    acc00 = __builtin_amdgcn_wmma_f32_16x16x32_bf16(false, a0h, false, b0l, (short)0, acc00, false, false);
    acc00 = __builtin_amdgcn_wmma_f32_16x16x32_bf16(false, a0l, false, b0h, (short)0, acc00, false, false);

    acc01 = __builtin_amdgcn_wmma_f32_16x16x32_bf16(false, a0h, false, b1h, (short)0, acc01, false, false);
    acc01 = __builtin_amdgcn_wmma_f32_16x16x32_bf16(false, a0h, false, b1l, (short)0, acc01, false, false);
    acc01 = __builtin_amdgcn_wmma_f32_16x16x32_bf16(false, a0l, false, b1h, (short)0, acc01, false, false);

    acc10 = __builtin_amdgcn_wmma_f32_16x16x32_bf16(false, a1h, false, b0h, (short)0, acc10, false, false);
    acc10 = __builtin_amdgcn_wmma_f32_16x16x32_bf16(false, a1h, false, b0l, (short)0, acc10, false, false);
    acc10 = __builtin_amdgcn_wmma_f32_16x16x32_bf16(false, a1l, false, b0h, (short)0, acc10, false, false);

    acc11 = __builtin_amdgcn_wmma_f32_16x16x32_bf16(false, a1h, false, b1h, (short)0, acc11, false, false);
    acc11 = __builtin_amdgcn_wmma_f32_16x16x32_bf16(false, a1h, false, b1l, (short)0, acc11, false, false);
    acc11 = __builtin_amdgcn_wmma_f32_16x16x32_bf16(false, a1l, false, b1h, (short)0, acc11, false, false);
  }

  // C/D layout: VGPR v: rows M = v (lanes 0-15) / v+8 (lanes 16-31), col = lane%16
  const int jc = lane & 15;
  const int ir = (lane < 16) ? 0 : 8;
  auto emit = [&](const v8f& acc, int ii0, int jj0) {
    const int jj = jj0 + jc;
    const float x2j = x2[jj];
#pragma unroll
    for (int v = 0; v < 8; ++v) {
      const int i = ii0 + ir + v;
      float val = x2[i] + x2j - 2.0f * acc[v];
      d2[(size_t)i * n + jj] = val;
      atomicAdd(&lh[sortable_key(val) >> 24], 1u);   // fused radix pass 1
    }
  };
  emit(acc00, i0,      j0);
  emit(acc01, i0,      j0 + 16);
  emit(acc10, i0 + 16, j0);
  emit(acc11, i0 + 16, j0 + 16);

  __syncthreads();
  unsigned c = lh[threadIdx.x];
  if (c) atomicAdd(&hist[threadIdx.x], c);
}

// ---------------- exact median: MSB-first radix select on sortable keys ------
__global__ void sel_init_kernel(unsigned* state, unsigned rank0) {
  state[0] = 0u;       // prefix
  state[1] = rank0;    // remaining rank
}
__global__ void hist_zero_kernel(unsigned* hist) { hist[threadIdx.x] = 0u; }

__global__ void hist_count_kernel(const float* __restrict__ v, unsigned total,
                                  const unsigned* __restrict__ state,
                                  int shift, unsigned mask,
                                  unsigned* __restrict__ hist) {
  __shared__ unsigned lh[256];
  lh[threadIdx.x] = 0u; __syncthreads();
  const unsigned prefix = state[0];
  const unsigned stride = gridDim.x * blockDim.x;
  for (unsigned idx = blockIdx.x * blockDim.x + threadIdx.x; idx < total; idx += stride) {
    unsigned u = sortable_key(v[idx]);
    if ((u & mask) == prefix) atomicAdd(&lh[(u >> shift) & 0xFFu], 1u);
  }
  __syncthreads();
  unsigned c = lh[threadIdx.x];
  if (c) atomicAdd(&hist[threadIdx.x], c);
}

__global__ void sel_scan_kernel(const unsigned* __restrict__ hist,
                                unsigned* __restrict__ state,
                                int shift, float* __restrict__ med) {
  unsigned rank = state[1];
  unsigned cum = 0, digit = 255;
  for (int b = 0; b < 256; ++b) {
    unsigned c = hist[b];
    if (cum + c > rank) { digit = (unsigned)b; break; }
    cum += c;
  }
  state[0] |= digit << shift;
  state[1] = rank - cum;
  if (shift == 0) {
    unsigned key = state[0];
    unsigned bits = (key & 0x80000000u) ? (key & 0x7FFFFFFFu) : ~key;
    *med = __uint_as_float(bits);
  }
}

// ---------------- row sums of K = exp(-d2/denom) ----------------
__global__ void rowsum_exp_kernel(const float* __restrict__ d2,
                                  const float* __restrict__ med,
                                  float* __restrict__ rowsum, int n) {
  __shared__ float sh[256];
  const int i = blockIdx.x;
  const float m = *med;
  const float sigma2 = 0.5f * m + 1e-8f;
  const float inv = -1.0f / (2.0f * sigma2 + 1e-8f);
  float s = 0.f;
  for (int j = threadIdx.x; j < n; j += 256)
    s += expf(d2[(size_t)i * n + j] * inv);
  sh[threadIdx.x] = s; __syncthreads();
  for (int off = 128; off > 0; off >>= 1) {
    if (threadIdx.x < off) sh[threadIdx.x] += sh[threadIdx.x + off];
    __syncthreads();
  }
  if (threadIdx.x == 0) rowsum[i] = sh[0];
}

__global__ void total_sum_kernel(const float* __restrict__ rowK,
                                 const float* __restrict__ rowL,
                                 float* __restrict__ scal, int n) {
  __shared__ float shk[256], shl[256];
  float sk = 0.f, sl = 0.f;
  for (int i = threadIdx.x; i < n; i += 256) { sk += rowK[i]; sl += rowL[i]; }
  shk[threadIdx.x] = sk; shl[threadIdx.x] = sl; __syncthreads();
  for (int off = 128; off > 0; off >>= 1) {
    if (threadIdx.x < off) { shk[threadIdx.x] += shk[threadIdx.x + off]; shl[threadIdx.x] += shl[threadIdx.x + off]; }
    __syncthreads();
  }
  if (threadIdx.x == 0) { scal[2] = shk[0]; scal[3] = shl[0]; }
}

// ---------------- fused final pass: sum(Kc * Lc) per row ----------------
// scal: [0]=medZ [1]=medN [2]=totK [3]=totL
__global__ void hsic_row_kernel(const float* __restrict__ d2Z,
                                const float* __restrict__ d2N,
                                const float* __restrict__ rowK,
                                const float* __restrict__ rowL,
                                const float* __restrict__ scal,
                                float* __restrict__ partial, int n) {
  __shared__ float sh[256];
  const int i = blockIdx.x;
  const float invZ = -1.0f / (2.0f * (0.5f * scal[0] + 1e-8f) + 1e-8f);
  const float invN = -1.0f / (2.0f * (0.5f * scal[1] + 1e-8f) + 1e-8f);
  const float rn = 1.0f / (float)n;
  const float tK = scal[2] * rn * rn;
  const float tL = scal[3] * rn * rn;
  const float aK = rowK[i] * rn;
  const float aL = rowL[i] * rn;
  float s = 0.f;
  for (int j = threadIdx.x; j < n; j += 256) {
    float Kij = expf(d2Z[(size_t)i * n + j] * invZ);
    float Lij = expf(d2N[(size_t)i * n + j] * invN);
    float kc = Kij - aK - rowK[j] * rn + tK;
    float lc = Lij - aL - rowL[j] * rn + tL;
    s += kc * lc;
  }
  sh[threadIdx.x] = s; __syncthreads();
  for (int off = 128; off > 0; off >>= 1) {
    if (threadIdx.x < off) sh[threadIdx.x] += sh[threadIdx.x + off];
    __syncthreads();
  }
  if (threadIdx.x == 0) partial[i] = sh[0];
}

__global__ void final_reduce_kernel(const float* __restrict__ partial,
                                    float* __restrict__ out, int n) {
  __shared__ float sh[256];
  float s = 0.f;
  for (int i = threadIdx.x; i < n; i += 256) s += partial[i];
  sh[threadIdx.x] = s; __syncthreads();
  for (int off = 128; off > 0; off >>= 1) {
    if (threadIdx.x < off) sh[threadIdx.x] += sh[threadIdx.x + off];
    __syncthreads();
  }
  if (threadIdx.x == 0) {
    float denom = (float)(n - 1) * (float)(n - 1) + 1e-8f;
    out[0] = sh[0] / denom;
  }
}

extern "C" void kernel_launch(void* const* d_in, const int* in_sizes, int n_in,
                              void* d_out, int out_size, void* d_ws, size_t ws_size,
                              hipStream_t stream) {
  const float* Z = (const float*)d_in[0];   // [4096, 512]
  const float* N = (const float*)d_in[1];   // [4096, 128]
  float* out = (float*)d_out;

  constexpr size_t n  = NPTS;
  constexpr size_t M2 = n * n;              // 16,777,216

  char* ws = (char*)d_ws;
  size_t off = 0;
  float*          d2Z   = (float*)(ws + off);          off += M2 * 4;         // 64 MiB
  float*          d2N   = (float*)(ws + off);          off += M2 * 4;         // 64 MiB
  unsigned short* Zhi   = (unsigned short*)(ws + off); off += n * DZ * 2;
  unsigned short* Zlo   = (unsigned short*)(ws + off); off += n * DZ * 2;
  unsigned short* Nhi   = (unsigned short*)(ws + off); off += n * DN * 2;
  unsigned short* Nlo   = (unsigned short*)(ws + off); off += n * DN * 2;
  float*          x2Z   = (float*)(ws + off);          off += n * 4;
  float*          x2N   = (float*)(ws + off);          off += n * 4;
  float*          rowK  = (float*)(ws + off);          off += n * 4;
  float*          rowL  = (float*)(ws + off);          off += n * 4;
  float*          part  = (float*)(ws + off);          off += n * 4;
  float*          scal  = (float*)(ws + off);          off += 256;            // [medZ, medN, totK, totL]
  unsigned*       stZ   = (unsigned*)(ws + off);       off += 256;
  unsigned*       stN   = (unsigned*)(ws + off);       off += 256;
  unsigned*       histZ = (unsigned*)(ws + off);       off += 256 * 4;
  unsigned*       histN = (unsigned*)(ws + off);       off += 256 * 4;
  (void)ws_size; (void)in_sizes; (void)n_in; (void)out_size;

  // 0) zero the fused pass-1 histograms
  hist_zero_kernel<<<1, 256, 0, stream>>>(histZ);
  hist_zero_kernel<<<1, 256, 0, stream>>>(histN);

  // 1) split to bf16 hi/lo + row squared norms
  split_bf16_kernel<<<(int)((n * DZ + 255) / 256), 256, 0, stream>>>(Z, Zhi, Zlo, (int)(n * DZ));
  split_bf16_kernel<<<(int)((n * DN + 255) / 256), 256, 0, stream>>>(N, Nhi, Nlo, (int)(n * DN));
  row_sqnorm_kernel<<<(int)n, 256, 0, stream>>>(Z, x2Z, DZ);
  row_sqnorm_kernel<<<(int)n, 256, 0, stream>>>(N, x2N, DN);

  // 2) d2 matrices via WMMA (2x2 tiles/wave, 8 waves/block) + fused histogram
  const int mtiles = (int)((n / 32) * (n / 32));  // 16384 macro tiles
  const int blocks = mtiles / 8;                  // 2048
  d2_wmma_hist_kernel<<<blocks, 256, 0, stream>>>(Zhi, Zlo, x2Z, d2Z, (int)n, DZ, histZ);
  d2_wmma_hist_kernel<<<blocks, 256, 0, stream>>>(Nhi, Nlo, x2N, d2N, (int)n, DN, histN);

  // 3) exact medians (lower-middle element): pass 1 fused above, 3 passes here
  const unsigned rank0 = (unsigned)((M2 - 1) / 2); // 8388607
  struct { const float* v; unsigned* st; unsigned* h; float* med; } sel[2] = {
    { d2Z, stZ, histZ, scal + 0 }, { d2N, stN, histN, scal + 1 } };
  for (int s = 0; s < 2; ++s) {
    sel_init_kernel<<<1, 1, 0, stream>>>(sel[s].st, rank0);
    sel_scan_kernel<<<1, 1, 0, stream>>>(sel[s].h, sel[s].st, 24, sel[s].med);
    for (int shift = 16; shift >= 0; shift -= 8) {
      unsigned mask = 0xFFFFFFFFu << (shift + 8);
      hist_zero_kernel<<<1, 256, 0, stream>>>(sel[s].h);
      hist_count_kernel<<<4096, 256, 0, stream>>>(sel[s].v, (unsigned)M2, sel[s].st, shift, mask, sel[s].h);
      sel_scan_kernel<<<1, 1, 0, stream>>>(sel[s].h, sel[s].st, shift, sel[s].med);
    }
  }

  // 4) centering row sums + totals
  rowsum_exp_kernel<<<(int)n, 256, 0, stream>>>(d2Z, scal + 0, rowK, (int)n);
  rowsum_exp_kernel<<<(int)n, 256, 0, stream>>>(d2N, scal + 1, rowL, (int)n);
  total_sum_kernel<<<1, 256, 0, stream>>>(rowK, rowL, scal, (int)n);

  // 5) fused sum(Kc*Lc) and final scalar
  hsic_row_kernel<<<(int)n, 256, 0, stream>>>(d2Z, d2N, rowK, rowL, scal, part, (int)n);
  final_reduce_kernel<<<1, 256, 0, stream>>>(part, out, (int)n);
}
